// Generator_88218628260353
// MI455X (gfx1250) — matrix operand order repaired and checked by reading, not compiled, not run on hardware
//
#include <hip/hip_runtime.h>
#include <hip/hip_bf16.h>

// Problem constants from the reference.
#define BB 16
#define TT 64
#define DD 512
#define SS 400
#define VV 50000
#define MM (BB * TT)          // 1024 rows
#define MTILES (MM / 16)      // 64
#define NSUB 5                // subtiles (16 cols) per wave tile -> 80 cols
#define NGROUPS (VV / (16 * NSUB))  // 50000/80 = 625, EXACT (no tail, no guards)

typedef __attribute__((ext_vector_type(16))) _Float16 v16h;
typedef __attribute__((ext_vector_type(8)))  _Float16 v8h;
typedef __attribute__((ext_vector_type(8)))  float    v8f;

// ---------------------------------------------------------------- converts
__global__ void convert_x_f16(const float* __restrict__ x, _Float16* __restrict__ xh) {
    int i = blockIdx.x * blockDim.x + threadIdx.x;
    if (i < MM * DD) xh[i] = (_Float16)x[i];
}

// Transpose-convert W_proj (D x V, row-major) -> Wt (V x D, f16), LDS tiled.
// One block per 16-wide V stripe (3125 blocks).
__global__ __launch_bounds__(256) void convert_w_t(const float* __restrict__ W,
                                                   _Float16* __restrict__ Wt) {
    __shared__ _Float16 tile[16][DD + 8];   // +8 halves pad to spread banks
    const int v0 = blockIdx.x * 16;
    for (int idx = threadIdx.x; idx < DD * 16; idx += 256) {
        int k = idx >> 4;
        int v = idx & 15;
        tile[v][k] = (_Float16)W[(size_t)k * VV + v0 + v];
    }
    __syncthreads();
    for (int idx = threadIdx.x; idx < DD * 16; idx += 256) {
        int v = idx >> 9;       // idx / 512
        int k = idx & (DD - 1); // idx % 512
        Wt[(size_t)(v0 + v) * DD + k] = tile[v][k];
    }
}

// ---------------------------------------------------------------- p_gen
__global__ void pgen_kernel(const float* __restrict__ x, const float* __restrict__ Wp,
                            const float* __restrict__ bp, float* __restrict__ pgen) {
    int wid  = (blockIdx.x * blockDim.x + threadIdx.x) >> 5;
    int lane = threadIdx.x & 31;
    if (wid >= MM) return;
    const float* xr = x + (size_t)wid * DD;
    float s = 0.f;
    for (int k = lane; k < DD; k += 32) s += xr[k] * Wp[k];
    for (int off = 16; off > 0; off >>= 1) s += __shfl_down(s, off, 32);
    if (lane == 0) pgen[wid] = 1.f / (1.f + __expf(-(s + bp[0])));
}

// -------------------------------------------------------- attn softmax*(1-pg)
__global__ __launch_bounds__(256) void attn_softmax(const float* __restrict__ attn,
                                                    const float* __restrict__ pgen,
                                                    float* __restrict__ attn_out) {
    const int row = blockIdx.x;               // b*T + t
    const float* p = attn + (size_t)row * SS;
    __shared__ float red[256];
    float m = -INFINITY;
    for (int s = threadIdx.x; s < SS; s += 256) m = fmaxf(m, p[s]);
    red[threadIdx.x] = m; __syncthreads();
    for (int off = 128; off > 0; off >>= 1) {
        if (threadIdx.x < off) red[threadIdx.x] = fmaxf(red[threadIdx.x], red[threadIdx.x + off]);
        __syncthreads();
    }
    m = red[0]; __syncthreads();
    float s = 0.f;
    for (int i = threadIdx.x; i < SS; i += 256) s += __expf(p[i] - m);
    red[threadIdx.x] = s; __syncthreads();
    for (int off = 128; off > 0; off >>= 1) {
        if (threadIdx.x < off) red[threadIdx.x] += red[threadIdx.x + off];
        __syncthreads();
    }
    const float w = (1.f - pgen[row]) / red[0];
    for (int i = threadIdx.x; i < SS; i += 256)
        attn_out[(size_t)row * SS + i] = __expf(p[i] - m) * w;
}

// ---------------------------------------------------------------- WMMA GEMM
// logits(M x V) = xh(M x D) * Wt(V x D)^T + bias.
// Per wave: 16(M) x 80(N) tile = 5 subtiles; 50000/80 = 625 exact -> zero
// control flow in the K-loop (EXEC all-ones throughout, as WMMA requires).
// j-outer / K-inner: one live accumulator (8 VGPRs) at a time so the
// scheduler has registers to pipeline loads several K-steps ahead instead of
// s_wait_loadcnt 0 before every WMMA.  A fragments re-read per j are L0 hits
// (same per-wave addresses); B columns are shared by all 8 waves of a block.
__global__ __launch_bounds__(256) void gemm_wmma(const _Float16* __restrict__ xh,
                                                 const _Float16* __restrict__ wt,
                                                 const float* __restrict__ bias,
                                                 float* __restrict__ out) {
    const int gtid = blockIdx.x * blockDim.x + threadIdx.x;
    const int wid  = gtid >> 5;
    const int lane = threadIdx.x & 31;
    if (wid >= MTILES * NGROUPS) return;          // exact grid; kept for safety

    const int mTile  = wid % MTILES;
    const int nGroup = wid / MTILES;
    const int hi  = lane >> 4;                    // half-wave select
    const int l15 = lane & 15;
    const int v0g = nGroup * (16 * NSUB);

    const _Float16* aRow = xh + (size_t)(mTile * 16 + l15) * DD;

#pragma unroll
    for (int j = 0; j < NSUB; ++j) {
        // Keep A loads local to this j iteration (block cross-j CSE that would
        // otherwise pin ~128 VGPRs of A fragments live across the whole kernel).
        asm volatile("" ::: "memory");

        const _Float16* bCol = wt + (size_t)(v0g + j * 16 + l15) * DD + 16 * hi;
        // Warm L0 for the next subtile's B column while we compute this one.
        const _Float16* bPref = (j + 1 < NSUB)
            ? wt + (size_t)(v0g + (j + 1) * 16 + l15) * DD
            : bCol;

        v8f acc;
#pragma unroll
        for (int r = 0; r < 8; ++r) acc[r] = 0.f;

#pragma unroll
        for (int k0 = 0; k0 < DD; k0 += 32) {
            // A fragment (16x32 f16): halves 0..15 <- K=k0+16*hi .. +15, contiguous
            v8h a0 = *reinterpret_cast<const v8h*>(aRow + k0 + 16 * hi);
            v8h a1 = *reinterpret_cast<const v8h*>(aRow + k0 + 16 * hi + 8);
            v16h a;
#pragma unroll
            for (int i = 0; i < 8; ++i) { a[i] = a0[i]; a[i + 8] = a1[i]; }

            // B fragment (32x16 f16): halves -> K = k0 .. k0+15 (+16*hi), contiguous
            v16h b = *reinterpret_cast<const v16h*>(bCol + k0);

            if (j + 1 < NSUB)
                __builtin_prefetch(bPref + k0, 0, 3);   // global_prefetch_b8

            acc = __builtin_amdgcn_wmma_f32_16x16x32_f16(
                false, a, false, b, (short)0, acc, false, false);
        }

        const int v = v0g + j * 16 + l15;
        const float bv = bias[v];
#pragma unroll
        for (int r = 0; r < 8; ++r) {
            const int row = mTile * 16 + r + 8 * hi;   // C/D layout
            out[(size_t)row * VV + v] = acc[r] + bv;
        }
    }
}

// --------------------------------------------------- per-row max / log-sum-exp
__global__ __launch_bounds__(256) void row_stats(const float* __restrict__ out,
                                                 float* __restrict__ rowmax,
                                                 float* __restrict__ rowlogsum) {
    const int row = blockIdx.x;
    const float* p = out + (size_t)row * VV;
    float m = -INFINITY, s = 0.f;
    for (int v = threadIdx.x; v < VV; v += 256) {
        float x = p[v];
        if (x > m) { s = s * __expf(m - x) + 1.f; m = x; }
        else       { s += __expf(x - m); }
    }
    __shared__ float ms[256], ss[256];
    ms[threadIdx.x] = m; ss[threadIdx.x] = s; __syncthreads();
    for (int off = 128; off > 0; off >>= 1) {
        if (threadIdx.x < off) {
            float m2 = ms[threadIdx.x + off], s2 = ss[threadIdx.x + off];
            float M  = fmaxf(ms[threadIdx.x], m2);
            ss[threadIdx.x] = ss[threadIdx.x] * __expf(ms[threadIdx.x] - M)
                            + s2 * __expf(m2 - M);
            ms[threadIdx.x] = M;
        }
        __syncthreads();
    }
    if (threadIdx.x == 0) { rowmax[row] = ms[0]; rowlogsum[row] = __logf(ss[0]); }
}

// ---------------------------------------------- in-place pg*softmax (probs)
__global__ void scale_probs(float* __restrict__ out, const float* __restrict__ rowmax,
                            const float* __restrict__ rowlogsum,
                            const float* __restrict__ pgen) {
    const int row = blockIdx.y;
    const float m = rowmax[row], ls = rowlogsum[row], pg = pgen[row];
    const size_t base = (size_t)row * VV;
    for (int v = blockIdx.x * blockDim.x + threadIdx.x; v < VV; v += gridDim.x * blockDim.x)
        out[base + v] = pg * __expf(out[base + v] - m - ls);
}

// ---------------------------------------------------------------- scatter add
__global__ void scatter_add(float* __restrict__ out, const float* __restrict__ attn_out,
                            const int* __restrict__ enc) {
    const int tid = blockIdx.x * blockDim.x + threadIdx.x;
    if (tid >= BB * TT * SS) return;
    const int s  = tid % SS;
    const int bt = tid / SS;
    const int b  = bt / TT;
    const int v  = enc[(size_t)b * SS + s];
    atomicAdd(&out[(size_t)bt * VV + v], attn_out[tid]);
}

// ---------------------------------------------------------------- final log
__global__ void log_inplace(float* __restrict__ out) {
    const size_t n = (size_t)MM * VV;
    for (size_t i = (size_t)blockIdx.x * blockDim.x + threadIdx.x; i < n;
         i += (size_t)gridDim.x * blockDim.x)
        out[i] = __logf(out[i]);
}

// ---------------------------------------------------------------- launcher
extern "C" void kernel_launch(void* const* d_in, const int* in_sizes, int n_in,
                              void* d_out, int out_size, void* d_ws, size_t ws_size,
                              hipStream_t stream) {
    const float* x    = (const float*)d_in[0];  // (B,T,D)
    const float* attn = (const float*)d_in[1];  // (B,T,S)
    const int*   enc  = (const int*)  d_in[2];  // (B,S)
    const float* Wp   = (const float*)d_in[3];  // (D,V)
    const float* bp   = (const float*)d_in[4];  // (V,)
    const float* Wg   = (const float*)d_in[5];  // (D,1)
    const float* bg   = (const float*)d_in[6];  // (1,)
    float* out = (float*)d_out;                 // (B,T,V) — staged as logits first

    // workspace layout
    char* ws = (char*)d_ws;
    size_t off = 0;
    _Float16* xh = (_Float16*)(ws + off); off += (size_t)MM * DD * sizeof(_Float16);
    off = (off + 255) & ~(size_t)255;
    _Float16* wt = (_Float16*)(ws + off); off += (size_t)VV * DD * sizeof(_Float16);
    off = (off + 255) & ~(size_t)255;
    float* pgen      = (float*)(ws + off); off += MM * sizeof(float);
    float* rowmax    = (float*)(ws + off); off += MM * sizeof(float);
    float* rowlogsum = (float*)(ws + off); off += MM * sizeof(float);
    float* attn_out  = (float*)(ws + off); off += (size_t)MM * SS * sizeof(float);

    convert_x_f16<<<(MM * DD + 255) / 256, 256, 0, stream>>>(x, xh);
    convert_w_t<<<VV / 16, 256, 0, stream>>>(Wp, wt);
    pgen_kernel<<<(MM * 32 + 255) / 256, 256, 0, stream>>>(x, Wg, bg, pgen);
    attn_softmax<<<MM, 256, 0, stream>>>(attn, pgen, attn_out);

    const int totalWaves = MTILES * NGROUPS;          // 40000
    gemm_wmma<<<(totalWaves + 7) / 8, 256, 0, stream>>>(xh, wt, bp, out);

    row_stats<<<MM, 256, 0, stream>>>(out, rowmax, rowlogsum);
    scale_probs<<<dim3((VV + 255) / 256, MM), 256, 0, stream>>>(out, rowmax, rowlogsum, pgen);
    scatter_add<<<(BB * TT * SS + 255) / 256, 256, 0, stream>>>(out, attn_out, enc);
    log_inplace<<<8192, 256, 0, stream>>>(out);
}
